// Spatial_MBConv_Block_45423574122912
// MI455X (gfx1250) — compile-verified, matrix operand order checked by reading, not run on hardware
//
#include <hip/hip_runtime.h>

#define EPSV 1e-5f

typedef __attribute__((ext_vector_type(16))) __bf16          v16bf;
typedef __attribute__((ext_vector_type(2)))  __bf16          v2bf;
typedef __attribute__((ext_vector_type(16))) unsigned short  v16u;
typedef __attribute__((ext_vector_type(8)))  unsigned short  v8u;
typedef __attribute__((ext_vector_type(8)))  float           v8f;

// single float -> bf16 (RNE); native cast lets clang pick the HW lowering
__device__ __forceinline__ unsigned short f2bfu(float f) {
  __bf16 h = (__bf16)f;
  return __builtin_bit_cast(unsigned short, h);
}
// pack two floats -> two bf16 in one dword (low = a, high = b)
__device__ __forceinline__ unsigned f2bf_pk(float a, float b) {
#if __has_builtin(__builtin_amdgcn_cvt_pk_bf16_f32)
  v2bf t = __builtin_amdgcn_cvt_pk_bf16_f32(a, b);
#else
  v2bf t; t[0] = (__bf16)a; t[1] = (__bf16)b;
#endif
  return __builtin_bit_cast(unsigned, t);
}
__device__ __forceinline__ float bfu2f(unsigned short h) {
  union { unsigned u; float f; } v; v.u = ((unsigned)h) << 16;
  return v.f;
}
// K index held by element e of a 16-bit A/B fragment, lane-half hl (ISA 7.12.2)
__device__ __forceinline__ int kmap(int e, int hl) {
  int r = e >> 1;
  int base = (r < 4) ? (r << 1) : (8 + (r << 1));
  return base + (hl << 3) + (e & 1);
}
// inverse: original k (0..31) -> fragment-order slot k' = hl*16 + e
// note: kperm(2m) and kperm(2m+1) are adjacent (even, odd) slots
__device__ __forceinline__ int kperm(int k) {
  int hl = (k >> 3) & 1;
  int kk = k - hl * 8;                  // in {0..7} U {16..23}
  int r  = (kk < 8) ? (kk >> 1) : ((kk >> 1) - 4);
  return hl * 16 + r * 2 + (k & 1);
}
// load 16 contiguous fragment-order ushorts (two ds_load_b128)
__device__ __forceinline__ v16u ldsfrag(const unsigned short* p) {
  union { v16u v; v8u h[2]; } u;
  u.h[0] = *(const v8u*)(p);
  u.h[1] = *(const v8u*)(p + 8);
  return u.v;
}
__device__ __forceinline__ v8f wmma_bf16(v16u a, v16u b, v8f c) {
  return __builtin_amdgcn_wmma_f32_16x16x32_bf16(
      false, __builtin_bit_cast(v16bf, a),
      false, __builtin_bit_cast(v16bf, b),
      (short)0, c, false, false);
}
__device__ __forceinline__ v8f vzero8() {
  v8f z = {0.f,0.f,0.f,0.f,0.f,0.f,0.f,0.f};
  return z;
}

// ---------------- Kernel 1: expand 1x1 conv + BN + ReLU -> H (bf16) --------
// H[n][o(256)][p(7500)] = relu(bn(sum_c w_up[o,c] * x[n,c,p]))
#define XROW 72   // 64 K-slots + pad, 144B rows (16B aligned, conflict-free)
__global__ __launch_bounds__(256) void k_expand(
    const float* __restrict__ x, const float* __restrict__ w_up,
    const float* __restrict__ gg, const float* __restrict__ bb,
    const float* __restrict__ mm, const float* __restrict__ vv,
    unsigned short* __restrict__ H)
{
  __shared__ __align__(16) unsigned short Xs[64 * XROW];    // [pp][k']
  __shared__ __align__(16) unsigned short Wl[256 * XROW];   // [o][k']
  __shared__ float sc[256], bi[256];
  const int tid = threadIdx.x;
  const int n   = blockIdx.y;
  const int p0  = blockIdx.x * 64;
  {
    float s = gg[tid] * rsqrtf(vv[tid] + EPSV);
    sc[tid] = s; bi[tid] = bb[tid] - mm[tid] * s;
  }
  // stage weights (256x64) permuted bf16, paired stores
  for (int i = 0; i < 32; ++i) {
    int pr  = tid + i * 256;            // pair id
    int lin = pr * 2;
    int o = lin >> 6, c = lin & 63;     // c even
    *(unsigned*)&Wl[o * XROW + (c & 32) + kperm(c & 31)] =
        f2bf_pk(w_up[lin], w_up[lin + 1]);
  }
  // stage x tile (64c x 64p) permuted bf16, pairs along c
  const float* xn = x + (size_t)n * 64 * 7500;
  for (int i = 0; i < 8; ++i) {
    int pr = tid + i * 256;             // 0..2047
    int pp = pr & 63, c = (pr >> 6) * 2;
    int p = p0 + pp;
    float va = 0.f, vb = 0.f;
    if (p < 7500) { va = xn[c * 7500 + p]; vb = xn[(c + 1) * 7500 + p]; }
    *(unsigned*)&Xs[pp * XROW + (c & 32) + kperm(c & 31)] = f2bf_pk(va, vb);
  }
  __syncthreads();
  const int wave = tid >> 5, lane = tid & 31;
  const int hl = lane >> 4, ln = lane & 15;
  unsigned short* Hn = H + (size_t)n * 256 * 7500;
  #pragma unroll
  for (int oi = 0; oi < 2; ++oi) {
    const int ot = wave + oi * 8;
    const int o  = ot * 16 + ln;
    const unsigned short* wbase = Wl + o * XROW + hl * 16;
    v16u a0 = ldsfrag(wbase);
    v16u a1 = ldsfrag(wbase + 32);
    #pragma unroll
    for (int pt = 0; pt < 4; ++pt) {
      const int pp = pt * 16 + ln;
      const unsigned short* xbase = Xs + pp * XROW + hl * 16;
      v16u b0 = ldsfrag(xbase);
      v16u b1 = ldsfrag(xbase + 32);
      v8f accA = wmma_bf16(a0, b0, vzero8());   // independent chains
      v8f accB = wmma_bf16(a1, b1, vzero8());
      const int p = p0 + pp;
      #pragma unroll
      for (int r = 0; r < 8; r += 2) {
        int o0 = ot * 16 + r + hl * 8, o1 = o0 + 1;
        float v0 = fmaxf((accA[r]     + accB[r])     * sc[o0] + bi[o0], 0.f);
        float v1 = fmaxf((accA[r + 1] + accB[r + 1]) * sc[o1] + bi[o1], 0.f);
        unsigned pk = f2bf_pk(v0, v1);
        if (p < 7500) {
          Hn[o0 * 7500 + p] = (unsigned short)pk;
          Hn[o1 * 7500 + p] = (unsigned short)(pk >> 16);
        }
      }
    }
  }
}

// ------------- Kernel 2: depthwise graph conv + BN + ReLU + SE sums --------
// H2[n][c][t*25+w] = relu(bn(sum_k w_dw[k*256+c] * sum_v H[n,(k*256+c)/3,t,v]*A[k,v,w]))
#define YROW 40   // 32 K-slots + pad, 80B rows
__global__ __launch_bounds__(256) void k_graph(
    const unsigned short* __restrict__ H, const float* __restrict__ A,
    const float* __restrict__ w_dw,
    const float* __restrict__ gg, const float* __restrict__ bb,
    const float* __restrict__ mm, const float* __restrict__ vv,
    unsigned short* __restrict__ H2, float* __restrict__ Ssum)
{
  __shared__ __align__(16) unsigned short Y[256 * YROW];     // [c][v']
  __shared__ __align__(16) unsigned short Am[3 * 32 * YROW]; // [k][w][v']
  __shared__ float sc[256], bi[256], ssum[256];
  const int tid = threadIdx.x;
  const int n   = blockIdx.y;
  const int t0  = blockIdx.x * 10;
  // stage A (transposed to [k][w][v'], zero-padded), paired stores
  for (int i = tid; i < 3 * 32 * (YROW / 2); i += 256) {
    int k = i / (32 * (YROW / 2)), rem = i % (32 * (YROW / 2));
    int w = rem / (YROW / 2), vp = (rem % (YROW / 2)) * 2;
    float va = 0.f, vb = 0.f;
    if (vp < 32 && w < 25) {
      int v0 = kmap(vp & 15, vp >> 4);       // vp even -> pair is v0, v0+1
      if (v0 < 25)     va = A[k * 625 + v0 * 25 + w];
      if (v0 + 1 < 25) vb = A[k * 625 + (v0 + 1) * 25 + w];
    }
    *(unsigned*)&Am[(k * 32 + w) * YROW + vp] = f2bf_pk(va, vb);
  }
  for (int i = tid; i < 256 * YROW; i += 256) Y[i] = 0;   // pad slots stay zero
  {
    float s = gg[tid] * rsqrtf(vv[tid] + EPSV);
    sc[tid] = s; bi[tid] = bb[tid] - mm[tid] * s; ssum[tid] = 0.f;
  }
  __syncthreads();
  const int wave = tid >> 5, lane = tid & 31;
  const int hl = lane >> 4, ln = lane & 15;
  const size_t nb = (size_t)n * 256;
  float se[4][8];                      // per-lane SE partial sums over t
  #pragma unroll
  for (int i = 0; i < 4; ++i)
    #pragma unroll
    for (int r = 0; r < 8; ++r) se[i][r] = 0.f;
  for (int t = t0; t < t0 + 10; ++t) {
    v8f acc[4];
    #pragma unroll
    for (int i = 0; i < 4; ++i) acc[i] = vzero8();
    for (int k = 0; k < 3; ++k) {
      __syncthreads();
      {
        const int c   = tid;
        const int ch  = k * 256 + c;
        const int src = ch / 3;
        const float wd = w_dw[ch];
        const unsigned short* hrow = H + (nb + src) * 7500 + t * 25;
        unsigned short* yrow = Y + c * YROW;
        #pragma unroll
        for (int v = 0; v < 24; v += 2) {
          float va = wd * bfu2f(hrow[v]);
          float vb = wd * bfu2f(hrow[v + 1]);
          *(unsigned*)&yrow[kperm(v)] = f2bf_pk(va, vb);
        }
        yrow[kperm(24)] = f2bfu(wd * bfu2f(hrow[24]));
      }
      __syncthreads();
      #pragma unroll
      for (int i = 0; i < 4; ++i) {
        const int tile = wave * 4 + i;
        const int ct = tile >> 1, wt = tile & 1;
        v16u a  = ldsfrag(Y + (ct * 16 + ln) * YROW + hl * 16);
        v16u bv = ldsfrag(Am + (k * 32 + wt * 16 + ln) * YROW + hl * 16);
        acc[i] = wmma_bf16(a, bv, acc[i]);
      }
    }
    #pragma unroll
    for (int i = 0; i < 4; ++i) {
      const int tile = wave * 4 + i;
      const int ct = tile >> 1, wt = tile & 1;
      const int w = wt * 16 + ln;
      if (w < 25) {
        #pragma unroll
        for (int r = 0; r < 8; r += 2) {
          int c0 = ct * 16 + r + hl * 8, c1 = c0 + 1;
          float v0 = fmaxf(acc[i][r]     * sc[c0] + bi[c0], 0.f);
          float v1 = fmaxf(acc[i][r + 1] * sc[c1] + bi[c1], 0.f);
          unsigned pk = f2bf_pk(v0, v1);
          H2[(nb + c0) * 7500 + t * 25 + w] = (unsigned short)pk;
          H2[(nb + c1) * 7500 + t * 25 + w] = (unsigned short)(pk >> 16);
          se[i][r]     += v0;
          se[i][r + 1] += v1;
        }
      }
    }
  }
  // flush per-lane SE sums once per block (w>=25 lanes contributed zeros)
  #pragma unroll
  for (int i = 0; i < 4; ++i) {
    const int tile = wave * 4 + i;
    const int ct = tile >> 1, wt = tile & 1;
    if (wt * 16 + ln < 25) {
      #pragma unroll
      for (int r = 0; r < 8; ++r)
        atomicAdd(&ssum[ct * 16 + r + hl * 8], se[i][r]);
    }
  }
  __syncthreads();
  atomicAdd(&Ssum[n * 256 + tid], ssum[tid]);
}

// ---------------- Kernel 3: SE squeeze-excite FCs (tiny) -------------------
__global__ __launch_bounds__(256) void k_se(
    const float* __restrict__ Ssum, const float* __restrict__ w1,
    const float* __restrict__ w2, float* __restrict__ G)
{
  __shared__ float ls[256], lz[64];
  const int tid = threadIdx.x;
  for (int n = 0; n < 32; ++n) {
    ls[tid] = Ssum[n * 256 + tid] * (1.f / 7500.f);
    __syncthreads();
    if (tid < 64) {
      float z = 0.f;
      for (int c = 0; c < 256; ++c) z += ls[c] * w1[tid * 256 + c];
      lz[tid] = fmaxf(z, 0.f);
    }
    __syncthreads();
    {
      float s = 0.f;
      for (int j = 0; j < 64; ++j) s += lz[j] * w2[tid * 64 + j];
      G[n * 256 + tid] = 1.f / (1.f + __expf(-s));
    }
    __syncthreads();
  }
}

// --------- Kernel 4: gate * project 1x1 + BN + residual + ReLU -> out ------
#define BROW 264  // 256 K-slots + pad, 528B rows (16B aligned, conflict-free)
__global__ __launch_bounds__(256) void k_project(
    const unsigned short* __restrict__ H2, const float* __restrict__ G,
    const float* __restrict__ w_down, const float* __restrict__ x,
    const float* __restrict__ gg, const float* __restrict__ bb,
    const float* __restrict__ mm, const float* __restrict__ vv,
    float* __restrict__ out)
{
  __shared__ __align__(16) unsigned short Bs[64 * BROW];  // [pp][k'] gated
  __shared__ __align__(16) unsigned short Wd[64 * BROW];  // [o][k']
  __shared__ float gl[256], scd[64], bid[64];
  const int tid = threadIdx.x;
  const int n   = blockIdx.y;
  const int p0  = blockIdx.x * 64;
  gl[tid] = G[n * 256 + tid];
  if (tid < 64) {
    float s = gg[tid] * rsqrtf(vv[tid] + EPSV);
    scd[tid] = s; bid[tid] = bb[tid] - mm[tid] * s;
  }
  __syncthreads();
  const unsigned short* H2n = H2 + (size_t)n * 256 * 7500;
  for (int i = 0; i < 32; ++i) {
    int pr = tid + i * 256;                  // pair id 0..8191
    // gated activations: pair along c (adjacent permuted slots)
    int pp = pr & 63, c = (pr >> 6) * 2;
    int p = p0 + pp;
    float va = 0.f, vb = 0.f;
    if (p < 7500) {
      va = bfu2f(H2n[c * 7500 + p]) * gl[c];
      vb = bfu2f(H2n[(c + 1) * 7500 + p]) * gl[c + 1];
    }
    *(unsigned*)&Bs[pp * BROW + (c & 0xE0) + kperm(c & 31)] = f2bf_pk(va, vb);
    // weights (64x256)
    int lin = pr * 2;
    int o2 = lin >> 8, c2 = lin & 255;       // c2 even
    *(unsigned*)&Wd[o2 * BROW + (c2 & 0xE0) + kperm(c2 & 31)] =
        f2bf_pk(w_down[lin], w_down[lin + 1]);
  }
  __syncthreads();
  const int wave = tid >> 5, lane = tid & 31;
  const int hl = lane >> 4, ln = lane & 15;
  const int ot  = wave >> 1;
  const int ptb = (wave & 1) * 2;
  const int o   = ot * 16 + ln;
  const unsigned short* wbase  = Wd + o * BROW + hl * 16;
  const unsigned short* bbase0 = Bs + (ptb * 16 + ln) * BROW + hl * 16;
  const unsigned short* bbase1 = bbase0 + 16 * BROW;
  v8f a00 = vzero8(), a01 = vzero8(), a10 = vzero8(), a11 = vzero8();
  #pragma unroll
  for (int ks = 0; ks < 8; ks += 2) {        // 4 independent chains
    v16u aE  = ldsfrag(wbase  + ks * 32);
    v16u b0E = ldsfrag(bbase0 + ks * 32);
    v16u b1E = ldsfrag(bbase1 + ks * 32);
    v16u aO  = ldsfrag(wbase  + ks * 32 + 32);
    v16u b0O = ldsfrag(bbase0 + ks * 32 + 32);
    v16u b1O = ldsfrag(bbase1 + ks * 32 + 32);
    a00 = wmma_bf16(aE, b0E, a00);
    a10 = wmma_bf16(aE, b1E, a10);
    a01 = wmma_bf16(aO, b0O, a01);
    a11 = wmma_bf16(aO, b1O, a11);
  }
  const float* xn = x + (size_t)n * 64 * 7500;
  float* on = out + (size_t)n * 64 * 7500;
  #pragma unroll
  for (int r = 0; r < 8; ++r) {
    int oo = ot * 16 + r + hl * 8;
    int p1 = p0 + ptb * 16 + ln;
    int p2 = p1 + 16;
    float s = scd[oo], bc = bid[oo];
    float acc0 = a00[r] + a01[r];
    float acc1 = a10[r] + a11[r];
    if (p1 < 7500) on[oo * 7500 + p1] = fmaxf(acc0 * s + bc + xn[oo * 7500 + p1], 0.f);
    if (p2 < 7500) on[oo * 7500 + p2] = fmaxf(acc1 * s + bc + xn[oo * 7500 + p2], 0.f);
  }
}

extern "C" void kernel_launch(void* const* d_in, const int* in_sizes, int n_in,
                              void* d_out, int out_size, void* d_ws, size_t ws_size,
                              hipStream_t stream) {
  (void)in_sizes; (void)n_in; (void)out_size; (void)ws_size;
  const float* x      = (const float*)d_in[0];
  const float* A      = (const float*)d_in[1];
  const float* w_up   = (const float*)d_in[2];
  const float* bug    = (const float*)d_in[3];
  const float* bub    = (const float*)d_in[4];
  const float* bum    = (const float*)d_in[5];
  const float* buv    = (const float*)d_in[6];
  const float* w_dw   = (const float*)d_in[7];
  const float* bg     = (const float*)d_in[8];
  const float* bbn    = (const float*)d_in[9];
  const float* bm     = (const float*)d_in[10];
  const float* bv     = (const float*)d_in[11];
  const float* se_w1  = (const float*)d_in[12];
  const float* se_w2  = (const float*)d_in[13];
  const float* w_down = (const float*)d_in[14];
  const float* bdg    = (const float*)d_in[15];
  const float* bdb    = (const float*)d_in[16];
  const float* bdm    = (const float*)d_in[17];
  const float* bdv    = (const float*)d_in[18];
  float* out = (float*)d_out;

  const size_t HE = (size_t)32 * 256 * 7500;          // 61.44M elems
  unsigned short* H  = (unsigned short*)d_ws;
  unsigned short* H2 = H + HE;
  float* S = (float*)(H2 + HE);
  float* G = S + 32 * 256;

  hipMemsetAsync(S, 0, 32 * 256 * sizeof(float), stream);
  k_expand <<<dim3(118, 32), 256, 0, stream>>>(x, w_up, bug, bub, bum, buv, H);
  k_graph  <<<dim3(30, 32),  256, 0, stream>>>(H, A, w_dw, bg, bbn, bm, bv, H2, S);
  k_se     <<<1, 256, 0, stream>>>(S, se_w1, se_w2, G);
  k_project<<<dim3(118, 32), 256, 0, stream>>>(H2, G, w_down, x, bdg, bdb, bdm, bdv, out);
}
